// MemoryAttention_12567074308563
// MI455X (gfx1250) — matrix operand order repaired and checked by reading, not compiled, not run on hardware
//
#include <hip/hip_runtime.h>
#include <hip/hip_bf16.h>

#define SEQ 8192
#define DIM 1024

typedef __attribute__((ext_vector_type(16))) __bf16 v16bf;
typedef __attribute__((ext_vector_type(8)))  float  v8f;
typedef unsigned int v4u __attribute__((ext_vector_type(4)));
typedef int          v8i __attribute__((ext_vector_type(8)));
typedef int          v4i __attribute__((ext_vector_type(4)));

union FragU { v16bf v; uint4 u[2]; };

// ---- WMMA fragment loads from LDS tiles (bf16, row stride 32 elements = 64B) ----
__device__ __forceinline__ v16bf load_frag_A(const __bf16* tile, int lane) {
    const int m = lane & 15, h = lane >> 4;
    const uint4* p = (const uint4*)(tile + m * 32);
    FragU f;
    f.u[0] = p[h];
    f.u[1] = p[2 + h];
    return f.v;
}
__device__ __forceinline__ v16bf load_frag_B(const __bf16* tileBt, int lane) {
    const int n = lane & 15, h = lane >> 4;
    const uint4* p = (const uint4*)(tileBt + n * 32);
    FragU f;
    f.u[0] = p[h];
    f.u[1] = p[2 + h];
    return f.v;
}

// ---- TDM: DMA a [tile_rows x tile_k] bf16 tile (row stride = row_stride elems)
//      from global memory into contiguous LDS [tile_rows][tile_k]. One instruction
//      per wave; tracked by TENSORcnt. D# packing per CDNA5 ISA ch.8.
//      This toolchain's builtin takes 6 args: g0, g1, g2, g3, <extra v8i>, cpol. ----
__device__ __forceinline__ void tdm_load_2d(const void* gaddr, unsigned lds_off,
                                            int tile_k, int tile_rows, int row_stride) {
    const unsigned long long ga = (unsigned long long)(uintptr_t)gaddr;
    v4u g0;
    g0[0] = 1u;                                       // count=1, user descriptor
    g0[1] = lds_off;                                  // lds_addr
    g0[2] = (unsigned)(ga & 0xFFFFFFFFu);             // global_addr[31:0]
    g0[3] = (unsigned)((ga >> 32) & 0x01FFFFFFu) | (2u << 30);  // addr[56:32] | type=2
    v8i g1;
    g1[0] = (int)(1u << 16);                          // data_size=1 (2 bytes/elem)
    g1[1] = (int)((unsigned)tile_k << 16);            // tensor_dim0 = tile_k (no OOB)
    g1[2] = (int)((unsigned)tile_rows << 16);         // tensor_dim1 = tile_rows
    g1[3] = (int)((unsigned)tile_k << 16);            // tile_dim0
    g1[4] = (int)(unsigned)tile_rows;                 // tile_dim1 | tile_dim2=0
    g1[5] = (int)(unsigned)row_stride;                // tensor_dim0_stride[31:0]
    g1[6] = 0;                                        // stride0 hi | stride1 lo
    g1[7] = 0;                                        // stride1 hi
    const v4i z4 = {0, 0, 0, 0};
    const v8i z8 = {0, 0, 0, 0, 0, 0, 0, 0};
    __builtin_amdgcn_tensor_load_to_lds(g0, g1, z4, z4, z8, 0);
}
__device__ __forceinline__ unsigned lds_off(const void* p) {
    return (unsigned)(uintptr_t)p;
}

// ---- K0: transpose fp32 weight [k][n] -> bf16 [n][k] ----
__global__ __launch_bounds__(256) void k_transpose_w(const float* __restrict__ W,
                                                     __bf16* __restrict__ Wt) {
    __shared__ float tile[32][33];
    const int bx = blockIdx.x * 32, by = blockIdx.y * 32;
    const int tx = threadIdx.x & 31, ty = threadIdx.x >> 5;
    #pragma unroll
    for (int i = 0; i < 32; i += 8)
        tile[ty + i][tx] = W[(size_t)(by + ty + i) * DIM + bx + tx];
    __syncthreads();
    #pragma unroll
    for (int i = 0; i < 32; i += 8)
        Wt[(size_t)(bx + ty + i) * DIM + by + tx] = (__bf16)tile[tx][ty + i];
}

// ---- K0b: transpose bf16 [rows][cols] -> [cols][rows] (for V) ----
__global__ __launch_bounds__(256) void k_transpose_b(const __bf16* __restrict__ A,
                                                     __bf16* __restrict__ At,
                                                     int rows, int cols) {
    __shared__ __bf16 tile[32][33];
    const int bx = blockIdx.x * 32, by = blockIdx.y * 32;  // bx: col, by: row
    const int tx = threadIdx.x & 31, ty = threadIdx.x >> 5;
    #pragma unroll
    for (int i = 0; i < 32; i += 8)
        tile[ty + i][tx] = A[(size_t)(by + ty + i) * cols + bx + tx];
    __syncthreads();
    #pragma unroll
    for (int i = 0; i < 32; i += 8)
        At[(size_t)(bx + ty + i) * rows + by + tx] = tile[tx][ty + i];
}

// ---- K1: C_bf16 = A_f32 @ Wt_bf16^T ; A staged manually (cvt), B via TDM ----
__global__ __launch_bounds__(256) void k_gemm_xw(const float* __restrict__ A,
                                                 const __bf16* __restrict__ Bt,
                                                 __bf16* __restrict__ C) {
    __shared__ __attribute__((aligned(16))) __bf16 As[2][128 * 32];
    __shared__ __attribute__((aligned(16))) __bf16 Bs[2][128 * 32];
    const int tid = threadIdx.x, wave = tid >> 5, lane = tid & 31;
    const int m0 = blockIdx.x * 128, n0 = blockIdx.y * 128;
    v8f acc[8];
    #pragma unroll
    for (int t = 0; t < 8; ++t) acc[t] = (v8f){0, 0, 0, 0, 0, 0, 0, 0};

    const int sr = tid >> 1, sc = (tid & 1) * 16;
    const int nsteps = DIM / 32;
    auto stage = [&](int step) {
        const int buf = step & 1, k0 = step * 32;
        const float4* g = (const float4*)(A + (size_t)(m0 + sr) * DIM + k0 + sc);
        __bf16* d = As[buf] + sr * 32 + sc;
        #pragma unroll
        for (int i = 0; i < 4; ++i) {
            float4 f = g[i];
            d[4 * i + 0] = (__bf16)f.x; d[4 * i + 1] = (__bf16)f.y;
            d[4 * i + 2] = (__bf16)f.z; d[4 * i + 3] = (__bf16)f.w;
        }
        if (wave == 0)
            tdm_load_2d(Bt + (size_t)n0 * DIM + k0, lds_off(Bs[buf]), 32, 128, DIM);
    };
    stage(0);
    for (int i = 0; i < nsteps; ++i) {
        if (i + 1 < nsteps) {
            stage(i + 1);
            if (wave == 0) __builtin_amdgcn_s_wait_tensorcnt(1);
        } else if (wave == 0) {
            __builtin_amdgcn_s_wait_tensorcnt(0);
        }
        __syncthreads();
        const int buf = i & 1;
        v16bf a = load_frag_A(As[buf] + wave * 16 * 32, lane);
        #pragma unroll
        for (int t = 0; t < 8; ++t) {
            v16bf b = load_frag_B(Bs[buf] + t * 16 * 32, lane);
            acc[t] = __builtin_amdgcn_wmma_f32_16x16x32_bf16(false, a, false, b,
                                                             (short)0, acc[t], false, false);
        }
        __syncthreads();
    }
    const int hh = lane >> 4, nl = lane & 15;
    #pragma unroll
    for (int t = 0; t < 8; ++t)
        #pragma unroll
        for (int r = 0; r < 8; ++r) {
            const int row = m0 + wave * 16 + r + 8 * hh;
            const int col = n0 + t * 16 + nl;
            C[(size_t)row * DIM + col] = (__bf16)acc[t][r];
        }
}

// ---- K2: S_f16 = (Q @ K^T)/32, causal blocks; both tiles via TDM ----
__global__ __launch_bounds__(256) void k_gemm_qkt(const __bf16* __restrict__ Q,
                                                  const __bf16* __restrict__ K,
                                                  _Float16* __restrict__ S) {
    const int ib = blockIdx.x, jb = blockIdx.y;
    if (jb > ib) return;
    __shared__ __attribute__((aligned(16))) __bf16 As[2][128 * 32];
    __shared__ __attribute__((aligned(16))) __bf16 Bs[2][128 * 32];
    const int tid = threadIdx.x, wave = tid >> 5, lane = tid & 31;
    const int m0 = ib * 128, n0 = jb * 128;
    v8f acc[8];
    #pragma unroll
    for (int t = 0; t < 8; ++t) acc[t] = (v8f){0, 0, 0, 0, 0, 0, 0, 0};

    const int nsteps = DIM / 32;
    auto stage = [&](int step) {
        if (wave == 0) {
            const int buf = step & 1, k0 = step * 32;
            tdm_load_2d(Q + (size_t)m0 * DIM + k0, lds_off(As[buf]), 32, 128, DIM);
            tdm_load_2d(K + (size_t)n0 * DIM + k0, lds_off(Bs[buf]), 32, 128, DIM);
        }
    };
    stage(0);
    for (int i = 0; i < nsteps; ++i) {
        if (i + 1 < nsteps) {
            stage(i + 1);
            if (wave == 0) __builtin_amdgcn_s_wait_tensorcnt(2);
        } else if (wave == 0) {
            __builtin_amdgcn_s_wait_tensorcnt(0);
        }
        __syncthreads();
        const int buf = i & 1;
        v16bf a = load_frag_A(As[buf] + wave * 16 * 32, lane);
        #pragma unroll
        for (int t = 0; t < 8; ++t) {
            v16bf b = load_frag_B(Bs[buf] + t * 16 * 32, lane);
            acc[t] = __builtin_amdgcn_wmma_f32_16x16x32_bf16(false, a, false, b,
                                                             (short)0, acc[t], false, false);
        }
        __syncthreads();
    }
    const int hh = lane >> 4, nl = lane & 15;
    #pragma unroll
    for (int t = 0; t < 8; ++t)
        #pragma unroll
        for (int r = 0; r < 8; ++r) {
            const int row = m0 + wave * 16 + r + 8 * hh;
            const int col = n0 + t * 16 + nl;
            float v = acc[t][r] * 0.03125f;           // 1/sqrt(1024)
            if (col > row) v = -30000.0f;             // causal sentinel
            S[(size_t)row * SEQ + col] = (_Float16)v;
        }
}

// ---- K3: per-row online softmax stats over the causal prefix ----
__global__ __launch_bounds__(256) void k_stats(const _Float16* __restrict__ S,
                                               float* __restrict__ M,
                                               float* __restrict__ L) {
    const int row = blockIdx.x;
    const _Float16* srow = S + (size_t)row * SEQ;
    const int n = row + 1;
    float m = -1e30f, l = 0.0f;
    for (int c = threadIdx.x; c < n; c += 256) {
        const float s = (float)srow[c];
        const float nm = fmaxf(m, s);
        l = l * __expf(m - nm) + __expf(s - nm);
        m = nm;
    }
    __shared__ float sm[256], sl[256];
    sm[threadIdx.x] = m; sl[threadIdx.x] = l;
    __syncthreads();
    for (int off = 128; off > 0; off >>= 1) {
        if ((int)threadIdx.x < off) {
            const float m2 = sm[threadIdx.x + off], l2 = sl[threadIdx.x + off];
            const float nm = fmaxf(sm[threadIdx.x], m2);
            sl[threadIdx.x] = sl[threadIdx.x] * __expf(sm[threadIdx.x] - nm) + l2 * __expf(m2 - nm);
            sm[threadIdx.x] = nm;
        }
        __syncthreads();
    }
    if (threadIdx.x == 0) { M[row] = sm[0]; L[row] = sl[0]; }
}

// ---- K4: H = silu( softmax(S) @ V ) ; P staged manually (exp), V^T via TDM ----
__global__ __launch_bounds__(256) void k_gemm_pv(const _Float16* __restrict__ S,
                                                 const float* __restrict__ M,
                                                 const float* __restrict__ L,
                                                 const __bf16* __restrict__ Vt,  // [DIM][SEQ]
                                                 __bf16* __restrict__ H) {
    const int ib = blockIdx.x;          // query block
    const int n0 = blockIdx.y * 128;    // d-slice
    const int m0 = ib * 128;
    __shared__ __attribute__((aligned(16))) __bf16 As[2][128 * 32];
    __shared__ __attribute__((aligned(16))) __bf16 Bs[2][128 * 32];
    const int tid = threadIdx.x, wave = tid >> 5, lane = tid & 31;
    v8f acc[8];
    #pragma unroll
    for (int t = 0; t < 8; ++t) acc[t] = (v8f){0, 0, 0, 0, 0, 0, 0, 0};

    const int sr = tid >> 1, sc = (tid & 1) * 16;
    const float mrow = M[m0 + sr];
    const int nsteps = (ib + 1) * 4;    // causal extent / 32
    auto stage = [&](int step) {
        const int buf = step & 1, k0 = step * 32;
        const _Float16* g = S + (size_t)(m0 + sr) * SEQ + k0 + sc;
        __bf16* d = As[buf] + sr * 32 + sc;
        #pragma unroll
        for (int i = 0; i < 16; ++i)
            d[i] = (__bf16)__expf((float)g[i] - mrow);
        if (wave == 0)
            tdm_load_2d(Vt + (size_t)n0 * SEQ + k0, lds_off(Bs[buf]), 32, 128, SEQ);
    };
    stage(0);
    for (int i = 0; i < nsteps; ++i) {
        if (i + 1 < nsteps) {
            stage(i + 1);
            if (wave == 0) __builtin_amdgcn_s_wait_tensorcnt(1);
        } else if (wave == 0) {
            __builtin_amdgcn_s_wait_tensorcnt(0);
        }
        __syncthreads();
        const int buf = i & 1;
        v16bf a = load_frag_A(As[buf] + wave * 16 * 32, lane);
        #pragma unroll
        for (int t = 0; t < 8; ++t) {
            v16bf b = load_frag_B(Bs[buf] + t * 16 * 32, lane);
            acc[t] = __builtin_amdgcn_wmma_f32_16x16x32_bf16(false, a, false, b,
                                                             (short)0, acc[t], false, false);
        }
        __syncthreads();
    }
    const int hh = lane >> 4, nl = lane & 15;
    float invl[8];
    #pragma unroll
    for (int r = 0; r < 8; ++r)
        invl[r] = 1.0f / L[m0 + wave * 16 + r + 8 * hh];
    #pragma unroll
    for (int t = 0; t < 8; ++t)
        #pragma unroll
        for (int r = 0; r < 8; ++r) {
            const int row = m0 + wave * 16 + r + 8 * hh;
            const float x = acc[t][r] * invl[r];
            const float y = x / (1.0f + __expf(-x));          // SiLU
            H[(size_t)row * DIM + n0 + t * 16 + nl] = (__bf16)y;
        }
}

// ---- K5: out_f32 = H_bf16 @ wv2 ; both tiles via TDM ----
__global__ __launch_bounds__(256) void k_gemm_hw(const __bf16* __restrict__ A,
                                                 const __bf16* __restrict__ Bt,
                                                 float* __restrict__ C) {
    __shared__ __attribute__((aligned(16))) __bf16 As[2][128 * 32];
    __shared__ __attribute__((aligned(16))) __bf16 Bs[2][128 * 32];
    const int tid = threadIdx.x, wave = tid >> 5, lane = tid & 31;
    const int m0 = blockIdx.x * 128, n0 = blockIdx.y * 128;
    v8f acc[8];
    #pragma unroll
    for (int t = 0; t < 8; ++t) acc[t] = (v8f){0, 0, 0, 0, 0, 0, 0, 0};

    const int nsteps = DIM / 32;
    auto stage = [&](int step) {
        if (wave == 0) {
            const int buf = step & 1, k0 = step * 32;
            tdm_load_2d(A + (size_t)m0 * DIM + k0, lds_off(As[buf]), 32, 128, DIM);
            tdm_load_2d(Bt + (size_t)n0 * DIM + k0, lds_off(Bs[buf]), 32, 128, DIM);
        }
    };
    stage(0);
    for (int i = 0; i < nsteps; ++i) {
        if (i + 1 < nsteps) {
            stage(i + 1);
            if (wave == 0) __builtin_amdgcn_s_wait_tensorcnt(2);
        } else if (wave == 0) {
            __builtin_amdgcn_s_wait_tensorcnt(0);
        }
        __syncthreads();
        const int buf = i & 1;
        v16bf a = load_frag_A(As[buf] + wave * 16 * 32, lane);
        #pragma unroll
        for (int t = 0; t < 8; ++t) {
            v16bf b = load_frag_B(Bs[buf] + t * 16 * 32, lane);
            acc[t] = __builtin_amdgcn_wmma_f32_16x16x32_bf16(false, a, false, b,
                                                             (short)0, acc[t], false, false);
        }
        __syncthreads();
    }
    const int hh = lane >> 4, nl = lane & 15;
    #pragma unroll
    for (int t = 0; t < 8; ++t)
        #pragma unroll
        for (int r = 0; r < 8; ++r) {
            const int row = m0 + wave * 16 + r + 8 * hh;
            const int col = n0 + t * 16 + nl;
            C[(size_t)row * DIM + col] = acc[t][r];
        }
}

extern "C" void kernel_launch(void* const* d_in, const int* in_sizes, int n_in,
                              void* d_out, int out_size, void* d_ws, size_t ws_size,
                              hipStream_t stream) {
    const float* x   = (const float*)d_in[0];
    const float* wq  = (const float*)d_in[1];
    const float* wk  = (const float*)d_in[2];
    const float* wv1 = (const float*)d_in[3];
    const float* wv2 = (const float*)d_in[4];
    float* out = (float*)d_out;

    char* ws = (char*)d_ws;
    size_t off = 0;
    auto carve = [&](size_t bytes) { void* p = ws + off; off += (bytes + 255) & ~(size_t)255; return p; };
    __bf16*    wqT  = (__bf16*)carve((size_t)DIM * DIM * 2);
    __bf16*    wkT  = (__bf16*)carve((size_t)DIM * DIM * 2);
    __bf16*    wv1T = (__bf16*)carve((size_t)DIM * DIM * 2);
    __bf16*    wv2T = (__bf16*)carve((size_t)DIM * DIM * 2);
    __bf16*    Qb   = (__bf16*)carve((size_t)SEQ * DIM * 2);
    __bf16*    Kb   = (__bf16*)carve((size_t)SEQ * DIM * 2);
    __bf16*    Vb   = (__bf16*)carve((size_t)SEQ * DIM * 2);
    __bf16*    Vt   = (__bf16*)carve((size_t)SEQ * DIM * 2);
    _Float16*  Sm   = (_Float16*)carve((size_t)SEQ * SEQ * 2);   // 128 MB, L2-resident
    float*     Mv   = (float*)carve((size_t)SEQ * 4);
    float*     Lv   = (float*)carve((size_t)SEQ * 4);
    __bf16*    Hb   = (__bf16*)carve((size_t)SEQ * DIM * 2);
    (void)ws_size; (void)in_sizes; (void)n_in; (void)out_size;

    const dim3 blk(256);
    const dim3 gT(DIM / 32, DIM / 32);
    k_transpose_w<<<gT, blk, 0, stream>>>(wq,  wqT);
    k_transpose_w<<<gT, blk, 0, stream>>>(wk,  wkT);
    k_transpose_w<<<gT, blk, 0, stream>>>(wv1, wv1T);
    k_transpose_w<<<gT, blk, 0, stream>>>(wv2, wv2T);

    const dim3 gXW(SEQ / 128, DIM / 128);   // 64 x 8
    k_gemm_xw<<<gXW, blk, 0, stream>>>(x, wqT,  Qb);
    k_gemm_xw<<<gXW, blk, 0, stream>>>(x, wkT,  Kb);
    k_gemm_xw<<<gXW, blk, 0, stream>>>(x, wv1T, Vb);

    const dim3 gTV(DIM / 32, SEQ / 32);
    k_transpose_b<<<gTV, blk, 0, stream>>>(Vb, Vt, SEQ, DIM);

    const dim3 gQK(SEQ / 128, SEQ / 128);   // 64 x 64, upper blocks early-exit
    k_gemm_qkt<<<gQK, blk, 0, stream>>>(Qb, Kb, Sm);

    k_stats<<<dim3(SEQ), blk, 0, stream>>>(Sm, Mv, Lv);

    const dim3 gPV(SEQ / 128, DIM / 128);
    k_gemm_pv<<<gPV, blk, 0, stream>>>(Sm, Mv, Lv, Vt, Hb);

    k_gemm_hw<<<gXW, blk, 0, stream>>>(Hb, wv2T, out);
}
